// LinkPrediction_47811575939205
// MI455X (gfx1250) — compile-verified, matrix operand order checked by reading, not run on hardware
//
#include <hip/hip_runtime.h>

#define N_NODES   200000
#define N_EDGES   400000
#define EMB_DIM   128
#define HID_DIM   24
#define N_TYPES   8
#define K_TOT     256                       // 2 * EMB_DIM
#define MAX_PAD   (N_EDGES + N_TYPES * 16)  // 400128, each bucket padded to /16
#define MAX_TILES (MAX_PAD / 16)            // 25008
#define WAVES_PER_BLK 4
#define TILES_PER_WAVE 4                    // 25008 = 1563 blk * 4 waves * 4 tiles

typedef __attribute__((ext_vector_type(16))) __bf16 v16bf;
typedef __attribute__((ext_vector_type(8)))  __bf16 v8bf;
typedef __attribute__((ext_vector_type(8)))  float  v8f;

// immediate-offset ds_swizzle (group-of-32 xor mode) on a float
template <int IMM>
static __device__ __forceinline__ float swz(float x) {
  return __int_as_float(__builtin_amdgcn_ds_swizzle(__float_as_int(x), IMM));
}
#define SWAP1 0x041f
#define SWAP2 0x081f
#define SWAP4 0x101f
#define SWAP8 0x201f

// ---------------- bucketing kernels ----------------
__global__ void init_kernel(int* counts, int* cursor, int* padPerm) {
  int i = blockIdx.x * blockDim.x + threadIdx.x;
  if (i < MAX_PAD) padPerm[i] = -1;
  if (i < N_TYPES) { counts[i] = 0; cursor[i] = 0; }
}

__global__ void hist_kernel(const int* __restrict__ etype, int* counts) {
  int e = blockIdx.x * blockDim.x + threadIdx.x;
  if (e < N_EDGES) atomicAdd(&counts[etype[e]], 1);
}

__global__ void prefix_kernel(const int* __restrict__ counts, int* padBase) {
  if (threadIdx.x == 0) {
    int acc = 0;
    for (int t = 0; t < N_TYPES; ++t) {
      padBase[t] = acc;
      acc += ((counts[t] + 15) >> 4) << 4;
    }
    padBase[N_TYPES] = acc;
  }
}

__global__ void scatter_kernel(const int* __restrict__ etype,
                               const int* __restrict__ padBase,
                               int* cursor, int* padPerm) {
  int e = blockIdx.x * blockDim.x + threadIdx.x;
  if (e < N_EDGES) {
    int t = etype[e];
    int pos = padBase[t] + atomicAdd(&cursor[t], 1);
    padPerm[pos] = e;
  }
}

// ---------------- pack W1 into per-lane WMMA-B operand layout ----------------
__global__ void packB_kernel(const float* __restrict__ W1, __bf16* __restrict__ Bp) {
  int idx = blockIdx.x * blockDim.x + threadIdx.x;     // 8*8*2*32 = 4096
  if (idx >= N_TYPES * 8 * 2 * 32) return;
  int lane = idx & 31;
  int nt   = (idx >> 5) & 1;
  int kt   = (idx >> 6) & 7;
  int t    = idx >> 9;
  int n    = nt * 16 + (lane & 15);
  int kb   = kt * 32 + (lane >> 4) * 8;
  const float* w1t = W1 + (long long)t * K_TOT * HID_DIM;
  __bf16* out = Bp + (long long)idx * 16;
  #pragma unroll
  for (int i = 0; i < 8; ++i) {
    out[i]     = (n < HID_DIM) ? (__bf16)w1t[(kb + i)      * HID_DIM + n] : (__bf16)0.f;
    out[8 + i] = (n < HID_DIM) ? (__bf16)w1t[(kb + 16 + i) * HID_DIM + n] : (__bf16)0.f;
  }
}

// ---------------- convert + swizzle embeddings to bf16 WMMA order ----------------
// Each 32-K block is reordered [0..7 | 16..23 | 8..15 | 24..31] so one lane's
// 16 A-values per K-step are a single contiguous 32B (v16bf) load.
__global__ void cvt_emb_kernel(const float* __restrict__ emb, __bf16* __restrict__ out) {
  int idx = blockIdx.x * blockDim.x + threadIdx.x;     // N_NODES*16 groups of 8
  if (idx >= N_NODES * 16) return;
  int n = idx >> 4, rem = idx & 15, b = rem >> 2, q = rem & 3;
  int d = ((q & 1) << 1) | (q >> 1);                   // 0->0, 1->2, 2->1, 3->3
  const float* src = emb + (unsigned)n * EMB_DIM + b * 32 + q * 8;
  const float4 f0 = *(const float4*)src;
  const float4 f1 = *(const float4*)(src + 4);
  v8bf o;
  o[0] = (__bf16)f0.x; o[1] = (__bf16)f0.y; o[2] = (__bf16)f0.z; o[3] = (__bf16)f0.w;
  o[4] = (__bf16)f1.x; o[5] = (__bf16)f1.y; o[6] = (__bf16)f1.z; o[7] = (__bf16)f1.w;
  *(v8bf*)(out + (unsigned)n * EMB_DIM + b * 32 + d * 8) = o;
}

// pack 16 gathered floats into the A operand (hardware v_cvt_pk) — fp32 fallback
static __device__ __forceinline__ v16bf packA(float4 f0, float4 f1,
                                              float4 f2, float4 f3) {
  v16bf a;
  a[0]  = (__bf16)f0.x; a[1]  = (__bf16)f0.y; a[2]  = (__bf16)f0.z; a[3]  = (__bf16)f0.w;
  a[4]  = (__bf16)f1.x; a[5]  = (__bf16)f1.y; a[6]  = (__bf16)f1.z; a[7]  = (__bf16)f1.w;
  a[8]  = (__bf16)f2.x; a[9]  = (__bf16)f2.y; a[10] = (__bf16)f2.z; a[11] = (__bf16)f2.w;
  a[12] = (__bf16)f3.x; a[13] = (__bf16)f3.y; a[14] = (__bf16)f3.z; a[15] = (__bf16)f3.w;
  return a;
}

template <bool BF>
static __device__ __forceinline__ v16bf loadA(const float* baseF, const __bf16* baseB,
                                              int kt) {
  if (BF) {
    return *(const v16bf*)(baseB + kt * 32);           // one 32B pre-swizzled chunk
  } else {
    const float4 f0 = *(const float4*)(baseF + kt * 32);
    const float4 f1 = *(const float4*)(baseF + kt * 32 + 4);
    const float4 f2 = *(const float4*)(baseF + kt * 32 + 16);
    const float4 f3 = *(const float4*)(baseF + kt * 32 + 20);
    return packA(f0, f1, f2, f3);
  }
}

// ---------------- main MoE scoring kernel ----------------
// One wave owns TILES_PER_WAVE consecutive 16-edge tiles; the 16 packed-B
// operand vectors stay resident in VGPRs across tiles (reloaded only on a
// type change, <= 8 times across the whole grid). The A stream is manually
// software-pipelined (load step k+1 before the WMMAs of step k).
template <bool BF>
__global__ __launch_bounds__(32 * WAVES_PER_BLK)
void moe_kernel(const float* __restrict__ emb, const __bf16* __restrict__ embB,
                const int* __restrict__ ei,
                const float* __restrict__ b1, const float* __restrict__ W2,
                const float* __restrict__ b2, const int* __restrict__ padBase,
                const int* __restrict__ padPerm, const v16bf* __restrict__ Bp,
                float* __restrict__ scores) {
  const int wave  = threadIdx.x >> 5;
  const int lane  = threadIdx.x & 31;
  const int half  = lane >> 4;
  const int l16   = lane & 15;
  const int tile0 = (blockIdx.x * WAVES_PER_BLK + wave) * TILES_PER_WAVE;

  int pb[N_TYPES + 1];
  #pragma unroll
  for (int u = 0; u <= N_TYPES; ++u) pb[u] = padBase[u];

  v16bf Breg[16];
  float w2a = 0.f, w2b = 0.f, b1a = 0.f, b1b = 0.f, bb2 = 0.f;
  int tcur = -1;

  const int  nA   = l16;
  const int  nB   = 16 + l16;
  const bool nBok = (nB < HID_DIM);
  const int  nBc  = nBok ? nB : nA;

  for (int it = 0; it < TILES_PER_WAVE; ++it) {
    const int tile = tile0 + it;
    const int base = tile * 16;

    int t = 0;
    #pragma unroll
    for (int u = 1; u <= N_TYPES; ++u)
      if (base >= pb[u]) t = u;
    if (t >= N_TYPES) t = 0;

    if (t != tcur) {  // wave-uniform: refresh resident B + scalars
      tcur = t;
      const v16bf* blane = Bp + (long long)t * 8 * 2 * 32 + lane;
      #pragma unroll
      for (int q = 0; q < 16; ++q) Breg[q] = blane[q * 32];
      w2a = W2[t * HID_DIM + nA];
      w2b = nBok ? W2[t * HID_DIM + nBc] : 0.f;
      b1a = b1[t * HID_DIM + nA];
      b1b = nBok ? b1[t * HID_DIM + nBc] : 0.f;
      bb2 = b2[t];
    }

    const int e  = padPerm[base + l16];
    const int ec = (e < 0) ? 0 : e;
    const int srcN = ei[ec];
    const int dstN = ei[N_EDGES + ec];
    // 32-bit offsets (fit: 200000*128 < 2^31) -> SADDR+voffset addressing
    const float*  fS = emb  + (unsigned)srcN * EMB_DIM + half * 8;
    const float*  fD = emb  + (unsigned)dstN * EMB_DIM + half * 8;
    const __bf16* bS = embB + (unsigned)srcN * EMB_DIM + half * 16;
    const __bf16* bD = embB + (unsigned)dstN * EMB_DIM + half * 16;

    v8f c0 = {}; v8f c1 = {};
    // steps 0..3 -> src row (K 0..127), steps 4..7 -> dst row (K 128..255)
    v16bf aCur = loadA<BF>(fS, bS, 0);
    #pragma unroll
    for (int step = 0; step < 8; ++step) {
      v16bf aNext = aCur;
      if (step < 7) {
        const int nk = step + 1;
        aNext = (nk < 4) ? loadA<BF>(fS, bS, nk) : loadA<BF>(fD, bD, nk - 4);
      }
      c0 = __builtin_amdgcn_wmma_f32_16x16x32_bf16(false, aCur, false,
                                                   Breg[step * 2 + 0],
                                                   (short)0, c0, false, false);
      c1 = __builtin_amdgcn_wmma_f32_16x16x32_bf16(false, aCur, false,
                                                   Breg[step * 2 + 1],
                                                   (short)0, c1, false, false);
      aCur = aNext;
    }

    // C/D layout: VGPR r holds (M = r + half*8, N = l16 [+16 for c1])
    float v[8];
    #pragma unroll
    for (int r = 0; r < 8; ++r) {
      float x0 = c0[r] + b1a; x0 = (x0 > 0.f) ? x0 : 0.01f * x0;  // LeakyReLU
      float x1 = c1[r] + b1b; x1 = (x1 > 0.f) ? x1 : 0.01f * x1;
      v[r] = x0 * w2a + x1 * w2b;
    }
    float w[4];
    #pragma unroll
    for (int j = 0; j < 4; ++j) {
      const float a0 = v[2 * j]     + swz<SWAP1>(v[2 * j]);
      const float a1 = v[2 * j + 1] + swz<SWAP1>(v[2 * j + 1]);
      w[j] = (l16 & 1) ? a1 : a0;
    }
    float x[2];
    #pragma unroll
    for (int j = 0; j < 2; ++j) {
      const float b0  = w[2 * j]     + swz<SWAP2>(w[2 * j]);
      const float b1_ = w[2 * j + 1] + swz<SWAP2>(w[2 * j + 1]);
      x[j] = (l16 & 2) ? b1_ : b0;
    }
    const float y0 = x[0] + swz<SWAP4>(x[0]);
    const float y1 = x[1] + swz<SWAP4>(x[1]);
    const float y  = (l16 & 4) ? y1 : y0;
    const float z  = y + swz<SWAP8>(y);

    const int e8 = __builtin_amdgcn_ds_swizzle(e, SWAP8);
    const int ee = half ? e8 : e;
    if ((l16 < 8) && (ee >= 0)) scores[ee] = z + bb2;
  }
}

// ---------------- launcher ----------------
extern "C" void kernel_launch(void* const* d_in, const int* in_sizes, int n_in,
                              void* d_out, int out_size, void* d_ws, size_t ws_size,
                              hipStream_t stream) {
  const float* emb = (const float*)d_in[0];
  const int*   ei  = (const int*)  d_in[1];
  const int*   et  = (const int*)  d_in[2];
  const float* W1  = (const float*)d_in[3];
  const float* b1  = (const float*)d_in[4];
  const float* W2  = (const float*)d_in[5];
  const float* b2  = (const float*)d_in[6];
  float* out    = (float*)d_out;
  float* scores = out;

  const size_t EMB_BYTES = (size_t)N_NODES * EMB_DIM * sizeof(__bf16);  // 51.2MB
  const size_t INT_BYTES = (size_t)(32 + MAX_PAD) * sizeof(int);
  const size_t BP_BYTES  = (size_t)N_TYPES * 8 * 2 * 32 * 16 * sizeof(__bf16);
  const bool useBf = ws_size >= EMB_BYTES + INT_BYTES + BP_BYTES;

  char* p = (char*)d_ws;
  __bf16* embB = nullptr;
  if (useBf) { embB = (__bf16*)p; p += EMB_BYTES; }
  int* ws      = (int*)p;
  int* counts  = ws;
  int* cursor  = ws + 8;
  int* padBase = ws + 16;
  int* padPerm = ws + 32;
  __bf16* Bp   = (__bf16*)(ws + 32 + MAX_PAD);   // 32B-aligned in both layouts

  init_kernel   <<<(MAX_PAD + 255) / 256, 256, 0, stream>>>(counts, cursor, padPerm);
  hist_kernel   <<<(N_EDGES + 255) / 256, 256, 0, stream>>>(et, counts);
  prefix_kernel <<<1, 32, 0, stream>>>(counts, padBase);
  scatter_kernel<<<(N_EDGES + 255) / 256, 256, 0, stream>>>(et, padBase, cursor, padPerm);
  packB_kernel  <<<(N_TYPES * 8 * 2 * 32 + 255) / 256, 256, 0, stream>>>(W1, Bp);

  const int nBlocks = (MAX_TILES / TILES_PER_WAVE) / WAVES_PER_BLK;  // 1563 exact
  if (useBf) {
    cvt_emb_kernel<<<(N_NODES * 16 + 255) / 256, 256, 0, stream>>>(emb, embB);
    moe_kernel<true><<<nBlocks, 32 * WAVES_PER_BLK, 0, stream>>>(
        emb, embB, ei, b1, W2, b2, padBase, padPerm, (const v16bf*)Bp, scores);
  } else {
    moe_kernel<false><<<nBlocks, 32 * WAVES_PER_BLK, 0, stream>>>(
        emb, embB, ei, b1, W2, b2, padBase, padPerm, (const v16bf*)Bp, scores);
  }

  hipMemcpyAsync(out + N_EDGES, emb, (size_t)N_NODES * EMB_DIM * sizeof(float),
                 hipMemcpyDeviceToDevice, stream);
}
